// TriangleAttention_16793322127891
// MI455X (gfx1250) — matrix operand order repaired and checked by reading, not compiled, not run on hardware
//
#include <hip/hip_runtime.h>

// ---------------- problem constants ----------------
#define S_    256
#define C_    128
#define H_    4
#define D_    32
#define NPAIR (S_ * S_)          // 65536 pair positions

// ---------------- vector types for WMMA ----------------
typedef __attribute__((ext_vector_type(16))) __bf16        v16bf;
typedef __attribute__((ext_vector_type(8)))  float         v8f;
typedef __attribute__((ext_vector_type(16))) unsigned short v16u;
typedef __attribute__((ext_vector_type(8)))  unsigned short v8u;

// ---------------- bf16 bit helpers ----------------
__device__ __forceinline__ unsigned short f2bf(float f) {
    unsigned int u = __builtin_bit_cast(unsigned int, f);
    unsigned int r = u + 0x7FFFu + ((u >> 16) & 1u);   // round-to-nearest-even
    return (unsigned short)(r >> 16);
}
__device__ __forceinline__ float bf2f(unsigned short s) {
    unsigned int u = ((unsigned int)s) << 16;
    return __builtin_bit_cast(float, u);
}

// B fragment: lane = column (l%16), K contiguous starting at sel*16 (this code's k order, same as A)
__device__ __forceinline__ v16bf load_b16(const unsigned short* p) {
    v16u u = *(const v16u*)p;
    return __builtin_bit_cast(v16bf, u);
}
// A fragment: lane = row (l%16); K contiguous starting at sel*16
__device__ __forceinline__ v16bf load_a16(const unsigned short* p, int sel) {
    v16u u = *(const v16u*)(p + sel * 16);
    return __builtin_bit_cast(v16bf, u);
}

__device__ __forceinline__ v8f wmma_bf16(v16bf a, v16bf b, v8f c) {
    return __builtin_amdgcn_wmma_f32_16x16x32_bf16(false, a, false, b, (short)0, c,
                                                   false, false);
}

// =====================================================================
// Kernel 1: convert weights to bf16.  wcat rows: [0,128)=wq [128,256)=wk
// [256,384)=wv [384,512)=wg; each row is the C=128 input dim (K-contig =>
// directly loadable as WMMA B fragments).  wo_bf is w_o [c][e] row-major.
// =====================================================================
__global__ void prep_weights_kernel(const float* __restrict__ wq,
                                    const float* __restrict__ wk,
                                    const float* __restrict__ wv,
                                    const float* __restrict__ wg,
                                    const float* __restrict__ wo,
                                    unsigned short* __restrict__ wcat,
                                    unsigned short* __restrict__ wo_bf) {
    int idx = blockIdx.x * blockDim.x + threadIdx.x;
    if (idx < 512 * C_) {
        int row = idx / C_;
        const float* src = (row < 128) ? wq : (row < 256) ? wk : (row < 384) ? wv : wg;
        wcat[idx] = f2bf(src[(row & 127) * C_ + (idx & (C_ - 1))]);
    } else {
        int j = idx - 512 * C_;
        if (j < C_ * C_) wo_bf[j] = f2bf(wo[j]);
    }
}

// =====================================================================
// Kernel 2: LayerNorm over C per pair + tri_bias[h, p] = h[p,:].w_bias[h,:]
// one block (128 threads) per pair position, LDS tree reductions.
// =====================================================================
__global__ __launch_bounds__(128) void ln_bias_kernel(
    const float* __restrict__ x, const float* __restrict__ ln_w,
    const float* __restrict__ ln_b, const float* __restrict__ w_bias,
    unsigned short* __restrict__ h_bf, float* __restrict__ bias_out) {
    __shared__ float red[128];
    int p = blockIdx.x;
    int t = threadIdx.x;
    float v = x[p * C_ + t];

    red[t] = v; __syncthreads();
    for (int s = 64; s > 0; s >>= 1) { if (t < s) red[t] += red[t + s]; __syncthreads(); }
    float mu = red[0] * (1.0f / C_);
    __syncthreads();

    float d = v - mu;
    red[t] = d * d; __syncthreads();
    for (int s = 64; s > 0; s >>= 1) { if (t < s) red[t] += red[t + s]; __syncthreads(); }
    float var = red[0] * (1.0f / C_);
    __syncthreads();

    float val = d * rsqrtf(var + 1e-5f) * ln_w[t] + ln_b[t];
    h_bf[p * C_ + t] = f2bf(val);

#pragma unroll
    for (int hh = 0; hh < H_; ++hh) {
        red[t] = val * w_bias[hh * C_ + t]; __syncthreads();
        for (int s = 64; s > 0; s >>= 1) { if (t < s) red[t] += red[t + s]; __syncthreads(); }
        if (t == 0) bias_out[hh * NPAIR + p] = red[0];
        __syncthreads();
    }
}

// =====================================================================
// Kernel 3: fused projection GEMM  [65536 x 128] x [128 x 512] (bf16 WMMA)
// Block = 8 waves; each wave owns one 16-wide N tile; A-tile staged in LDS.
// `which` is derived from blockIdx -> uniform scalar branch (no exec-mask
// cascade).  Q is pre-scaled by 1/sqrt(D); V is written transposed; g gets
// sigmoid via single v_rcp_f32.
// =====================================================================
__global__ __launch_bounds__(256) void proj_kernel(
    const unsigned short* __restrict__ h_bf, const unsigned short* __restrict__ wcat,
    unsigned short* __restrict__ q_bf, unsigned short* __restrict__ k_bf,
    unsigned short* __restrict__ vT_bf, unsigned short* __restrict__ g_bf) {
    __shared__ unsigned short At[16 * C_];
    int bid = blockIdx.x;
    int m0 = (bid >> 2) * 16;
    int which = bid & 3;              // 0:q 1:k 2:v 3:g  (block-uniform, scalar)
    int tid = threadIdx.x;
    int w = tid >> 5, l = tid & 31;
    int sel = l >> 4, ln = l & 15;

    { // cooperative A tile load: 16 rows x 128 cols bf16
        int idx = tid * 8;
        int row = idx >> 7, col = idx & 127;
        *(v8u*)&At[idx] = *(const v8u*)&h_bf[(m0 + row) * C_ + col];
    }
    __syncthreads();

    int n0 = which * 128 + w * 16;
    v8f acc = {};
#pragma unroll
    for (int kc = 0; kc < 4; ++kc) {
        v16bf a = load_a16(&At[ln * C_ + kc * 32], sel);
        v16bf b = load_b16(&wcat[(n0 + ln) * C_ + kc * 32 + sel * 16]);
        acc = wmma_bf16(a, b, acc);
    }

    const float qscale = 0.17677669529663687f;    // 1/sqrt(D), folded into Q
    if (which == 0) {
#pragma unroll
        for (int r = 0; r < 8; ++r) {
            int m = m0 + r + sel * 8;
            int e = w * 16 + ln;
            q_bf[m * C_ + e] = f2bf(acc[r] * qscale);
        }
    } else if (which == 1) {
#pragma unroll
        for (int r = 0; r < 8; ++r) {
            int m = m0 + r + sel * 8;
            int e = w * 16 + ln;
            k_bf[m * C_ + e] = f2bf(acc[r]);
        }
    } else if (which == 2) {
#pragma unroll
        for (int r = 0; r < 8; ++r) {
            int m = m0 + r + sel * 8;
            int e = w * 16 + ln;
            int hh = e >> 5, d = e & 31;
            int i = m >> 8, ks = m & (S_ - 1);
            vT_bf[(((hh * S_) + i) * D_ + d) * S_ + ks] = f2bf(acc[r]);
        }
    } else {
#pragma unroll
        for (int r = 0; r < 8; ++r) {
            int m = m0 + r + sel * 8;
            int e = w * 16 + ln;
            float sg = __builtin_amdgcn_rcpf(1.0f + __expf(-acc[r]));
            g_bf[m * C_ + e] = f2bf(sg);
        }
    }
}

// =====================================================================
// Kernel 4: flash attention per (head, row i, j-half).  Each wave owns a
// 16-row j tile, loops over k in 32-wide chunks: 2 WMMA (QK^T) + online
// softmax (shfl_xor-16 row reductions match the wave32 C-layout halves) +
// P staged via per-wave LDS + 2 WMMA (P.V).  Fuses 1/l and sigmoid gate.
// =====================================================================
__global__ __launch_bounds__(256) void attn_kernel(
    const unsigned short* __restrict__ q_bf, const unsigned short* __restrict__ k_bf,
    const unsigned short* __restrict__ vT_bf, const unsigned short* __restrict__ g_bf,
    const float* __restrict__ bias, unsigned short* __restrict__ og_bf) {
    __shared__ unsigned short P[8][16 * 32];
    int bid = blockIdx.x;
    int jh = bid & 1;
    int i  = (bid >> 1) & (S_ - 1);
    int hh = bid >> 9;
    int tid = threadIdx.x;
    int w = tid >> 5, l = tid & 31;
    int sel = l >> 4, ln = l & 15;
    int j0 = jh * 128 + w * 16;

    v16bf aQ = load_a16(&q_bf[(i * S_ + j0 + ln) * C_ + hh * D_], sel);

    float mrow[8], lrow[8];
    v8f o0 = {}, o1 = {};
#pragma unroll
    for (int r = 0; r < 8; ++r) { mrow[r] = -3.0e38f; lrow[r] = 0.0f; }

    const float* bh = &bias[hh * NPAIR];
    const unsigned short* kbase = &k_bf[(i * S_) * C_ + hh * D_ + sel * 16];
    const unsigned short* vbase = &vT_bf[(((hh * S_) + i) * D_) * S_ + sel * 16];

#pragma unroll 2
    for (int kc = 0; kc < 8; ++kc) {
        int k0 = kc * 32;
        // prefetch next chunk's K / V^T lines into cache (global_prefetch_b8)
        if (kc < 7) {
            __builtin_prefetch(kbase + (k0 + 32 + ln) * C_, 0, 0);
            __builtin_prefetch(vbase + ln * S_ + k0 + 32, 0, 0);
        }
        v16bf b1 = load_b16(kbase + (k0 + ln) * C_);
        v16bf b2 = load_b16(kbase + (k0 + 16 + ln) * C_);
        v8f z = {};
        v8f s1 = wmma_bf16(aQ, b1, z);
        v8f s2 = wmma_bf16(aQ, b2, z);

#pragma unroll
        for (int r = 0; r < 8; ++r) {
            int j = j0 + r + sel * 8;
            float x1 = s1[r] + bh[j * S_ + k0 + ln];        // Q pre-scaled
            float x2 = s2[r] + bh[j * S_ + k0 + 16 + ln];
            float mx = fmaxf(x1, x2);
            mx = fmaxf(mx, __shfl_xor(mx, 1, 16));
            mx = fmaxf(mx, __shfl_xor(mx, 2, 16));
            mx = fmaxf(mx, __shfl_xor(mx, 4, 16));
            mx = fmaxf(mx, __shfl_xor(mx, 8, 16));
            float mnew  = fmaxf(mrow[r], mx);
            float alpha = __expf(mrow[r] - mnew);
            mrow[r] = mnew;
            float p1 = __expf(x1 - mnew);
            float p2 = __expf(x2 - mnew);
            float rs = p1 + p2;
            rs += __shfl_xor(rs, 1, 16);
            rs += __shfl_xor(rs, 2, 16);
            rs += __shfl_xor(rs, 4, 16);
            rs += __shfl_xor(rs, 8, 16);
            lrow[r] = lrow[r] * alpha + rs;
            o0[r] *= alpha;
            o1[r] *= alpha;
            int ml = r + sel * 8;                 // C-layout row
            P[w][ml * 32 + ln]      = f2bf(p1);
            P[w][ml * 32 + 16 + ln] = f2bf(p2);
        }

        v16bf aP  = load_a16(&P[w][ln * 32], sel);
        v16bf bv0 = load_b16(vbase + ln * S_ + k0);
        v16bf bv1 = load_b16(vbase + (16 + ln) * S_ + k0);
        o0 = wmma_bf16(aP, bv0, o0);
        o1 = wmma_bf16(aP, bv1, o1);
    }

#pragma unroll
    for (int r = 0; r < 8; ++r) {
        int j = j0 + r + sel * 8;
        float inv = __builtin_amdgcn_rcpf(lrow[r]);
        int base = (i * S_ + j) * C_ + hh * D_;
        float g0 = bf2f(g_bf[base + ln]);
        float g1 = bf2f(g_bf[base + 16 + ln]);
        og_bf[base + ln]      = f2bf(o0[r] * inv * g0);
        og_bf[base + 16 + ln] = f2bf(o1[r] * inv * g1);
    }
}

// =====================================================================
// Kernel 5: output projection  out[m,c] = sum_e og[m,e] * w_o[c,e]
// [65536 x 128] x [128 x 128] bf16 WMMA, fp32 result straight to d_out.
// =====================================================================
__global__ __launch_bounds__(256) void out_kernel(
    const unsigned short* __restrict__ og_bf, const unsigned short* __restrict__ wo_bf,
    float* __restrict__ out) {
    __shared__ unsigned short At[16 * C_];
    int m0 = blockIdx.x * 16;
    int tid = threadIdx.x;
    int w = tid >> 5, l = tid & 31;
    int sel = l >> 4, ln = l & 15;

    {
        int idx = tid * 8;
        int row = idx >> 7, col = idx & 127;
        *(v8u*)&At[idx] = *(const v8u*)&og_bf[(m0 + row) * C_ + col];
    }
    __syncthreads();

    int n0 = w * 16;
    v8f acc = {};
#pragma unroll
    for (int kc = 0; kc < 4; ++kc) {
        v16bf a = load_a16(&At[ln * C_ + kc * 32], sel);
        v16bf b = load_b16(&wo_bf[(n0 + ln) * C_ + kc * 32 + sel * 16]);
        acc = wmma_bf16(a, b, acc);
    }
#pragma unroll
    for (int r = 0; r < 8; ++r) {
        int m = m0 + r + sel * 8;
        out[m * C_ + n0 + ln] = acc[r];
    }
}

// =====================================================================
// Host-side launch
// =====================================================================
extern "C" void kernel_launch(void* const* d_in, const int* in_sizes, int n_in,
                              void* d_out, int out_size, void* d_ws, size_t ws_size,
                              hipStream_t stream) {
    (void)in_sizes; (void)n_in; (void)out_size; (void)ws_size;
    const float* x     = (const float*)d_in[0];
    const float* ln_w  = (const float*)d_in[1];
    const float* ln_b  = (const float*)d_in[2];
    const float* wbias = (const float*)d_in[3];
    const float* wq    = (const float*)d_in[4];
    const float* wk    = (const float*)d_in[5];
    const float* wv    = (const float*)d_in[6];
    const float* wg    = (const float*)d_in[7];
    const float* wo    = (const float*)d_in[8];

    char* ws = (char*)d_ws;
    size_t off = 0;
    auto carve = [&](size_t bytes) -> void* {
        void* p = ws + off;
        off += (bytes + 255) & ~size_t(255);
        return p;
    };
    unsigned short* h_bf  = (unsigned short*)carve((size_t)NPAIR * C_ * 2);       // 16 MB
    unsigned short* q_bf  = (unsigned short*)carve((size_t)NPAIR * C_ * 2);       // 16 MB
    unsigned short* k_bf  = (unsigned short*)carve((size_t)NPAIR * C_ * 2);       // 16 MB
    unsigned short* vT_bf = (unsigned short*)carve((size_t)H_ * S_ * D_ * S_ * 2);// 16 MB
    unsigned short* g_bf  = (unsigned short*)carve((size_t)NPAIR * C_ * 2);       // 16 MB
    unsigned short* og_bf = (unsigned short*)carve((size_t)NPAIR * C_ * 2);       // 16 MB
    float*          bias  = (float*)carve((size_t)H_ * NPAIR * 4);                //  1 MB
    unsigned short* wcat  = (unsigned short*)carve((size_t)512 * C_ * 2);         // 128 KB
    unsigned short* wo_bf = (unsigned short*)carve((size_t)C_ * C_ * 2);          //  32 KB

    // 1) weights -> bf16
    prep_weights_kernel<<<(512 * C_ + C_ * C_ + 255) / 256, 256, 0, stream>>>(
        wq, wk, wv, wg, wo, wcat, wo_bf);

    // 2) LayerNorm + tri-bias
    ln_bias_kernel<<<NPAIR, 128, 0, stream>>>(x, ln_w, ln_b, wbias, h_bf, bias);

    // 3) fused q/k/v/g projection GEMM
    proj_kernel<<<(NPAIR / 16) * 4, 256, 0, stream>>>(h_bf, wcat, q_bf, k_bf, vT_bf, g_bf);

    // 4) flash triangle attention + gating
    attn_kernel<<<H_ * S_ * 2, 256, 0, stream>>>(q_bf, k_bf, vT_bf, g_bf, bias, og_bf);

    // 5) output projection -> fp32 d_out
    out_kernel<<<NPAIR / 16, 256, 0, stream>>>(og_bf, wo_bf, (float*)d_out);
}
